// RadiologySelectionLoss_78958678770097
// MI455X (gfx1250) — compile-verified, matrix operand order checked by reading, not run on hardware
//
#include <hip/hip_runtime.h>

typedef __attribute__((ext_vector_type(16))) __bf16 v16bf;
typedef __attribute__((ext_vector_type(8)))  float  v8f;

#define TILE 128
#define KC 32        // K per WMMA step / per LDS chunk
#define PITCHB 40    // bf16 elements per LDS row: 80B rows, 16B-aligned, conflict-free

union FragBF {
    uint4 q[2];      // two 16B halves: K run [koff0,koff0+8) and [koff0+16,koff0+24)
    v16bf v;
};

__device__ __forceinline__ unsigned short f32_to_bf16_rne(float x) {
    unsigned u = __float_as_uint(x);
    unsigned r = u + 0x7FFFu + ((u >> 16) & 1u);
    return (unsigned short)(r >> 16);
}
__device__ __forceinline__ float bf16_bits_to_f32(unsigned short h) {
    return __uint_as_float(((unsigned)h) << 16);
}

__device__ __forceinline__ unsigned f2ord(float f) {
    unsigned u = __float_as_uint(f);
    return (u & 0x80000000u) ? ~u : (u | 0x80000000u);
}
__device__ __forceinline__ float ord2f(unsigned u) {
    return (u & 0x80000000u) ? __uint_as_float(u & 0x7fffffffu) : __uint_as_float(~u);
}

__global__ void radsel_init_ws(unsigned* __restrict__ p, int n4) {
    int i = blockIdx.x * blockDim.x + threadIdx.x;
    if (i < n4) p[i] = 0u;  // 0 is identity for ordered-max and for float sums
}

// PASS 1: row/col maxima of masked logits.  PASS 2: row/col sum of exp(x-max) + diag.
template <int PASS>
__global__ __launch_bounds__(256) void radsel_tile_kernel(
    const float* __restrict__ F, const float* __restrict__ I,
    const int* __restrict__ lab,
    unsigned* __restrict__ rmaxo, unsigned* __restrict__ cmaxo,
    float* __restrict__ rsum, float* __restrict__ csum,
    float* __restrict__ diag, int n, int d)
{
    // bf16 hi/lo split tiles: x == hi + lo to ~2^-17 relative accuracy
    __shared__ __align__(16) unsigned short Ahi[TILE * PITCHB];
    __shared__ __align__(16) unsigned short Alo[TILE * PITCHB];
    __shared__ __align__(16) unsigned short Bhi[TILE * PITCHB];
    __shared__ __align__(16) unsigned short Blo[TILE * PITCHB];

    const int tid  = threadIdx.x;
    const int lane = tid & 31;
    const int wave = tid >> 5;
    const int l16  = lane & 15;
    const int half = lane >> 4;
    const int koff0 = half * 8;          // lanes 0-15: K 0..7 & 16..23 ; 16-31: 8..15 & 24..31
    const int rowBase = blockIdx.y * TILE;
    const int colBase = blockIdx.x * TILE;

    v8f acc[8];
#pragma unroll
    for (int ct = 0; ct < 8; ++ct) acc[ct] = (v8f){0.f,0.f,0.f,0.f,0.f,0.f,0.f,0.f};

    // ---- tiled GEMM: acc[ct] = (F tile) x (I tile)^T via bf16x3 WMMA ----
    for (int kc = 0; kc < d; kc += KC) {
        __syncthreads();
#pragma unroll
        for (int r = 0; r < 4; ++r) {
            int idx = tid + r * 256;                 // 0..1023 float4 slots per matrix
            int row = idx >> 3, kq = idx & 7;
            float4 av = *(const float4*)(F + (size_t)(rowBase + row) * d + kc + kq * 4);
            float4 bv = *(const float4*)(I + (size_t)(colBase + row) * d + kc + kq * 4);
            uint2 ah, al, bh, bl;
            {
                unsigned short h0 = f32_to_bf16_rne(av.x), h1 = f32_to_bf16_rne(av.y);
                unsigned short h2 = f32_to_bf16_rne(av.z), h3 = f32_to_bf16_rne(av.w);
                unsigned short l0 = f32_to_bf16_rne(av.x - bf16_bits_to_f32(h0));
                unsigned short l1 = f32_to_bf16_rne(av.y - bf16_bits_to_f32(h1));
                unsigned short l2 = f32_to_bf16_rne(av.z - bf16_bits_to_f32(h2));
                unsigned short l3 = f32_to_bf16_rne(av.w - bf16_bits_to_f32(h3));
                ah = make_uint2((unsigned)h0 | ((unsigned)h1 << 16),
                                (unsigned)h2 | ((unsigned)h3 << 16));
                al = make_uint2((unsigned)l0 | ((unsigned)l1 << 16),
                                (unsigned)l2 | ((unsigned)l3 << 16));
            }
            {
                unsigned short h0 = f32_to_bf16_rne(bv.x), h1 = f32_to_bf16_rne(bv.y);
                unsigned short h2 = f32_to_bf16_rne(bv.z), h3 = f32_to_bf16_rne(bv.w);
                unsigned short l0 = f32_to_bf16_rne(bv.x - bf16_bits_to_f32(h0));
                unsigned short l1 = f32_to_bf16_rne(bv.y - bf16_bits_to_f32(h1));
                unsigned short l2 = f32_to_bf16_rne(bv.z - bf16_bits_to_f32(h2));
                unsigned short l3 = f32_to_bf16_rne(bv.w - bf16_bits_to_f32(h3));
                bh = make_uint2((unsigned)h0 | ((unsigned)h1 << 16),
                                (unsigned)h2 | ((unsigned)h3 << 16));
                bl = make_uint2((unsigned)l0 | ((unsigned)l1 << 16),
                                (unsigned)l2 | ((unsigned)l3 << 16));
            }
            const int o = row * PITCHB + kq * 4;
            *(uint2*)(&Ahi[o]) = ah;  *(uint2*)(&Alo[o]) = al;
            *(uint2*)(&Bhi[o]) = bh;  *(uint2*)(&Blo[o]) = bl;
        }
        __syncthreads();

        // A fragments for this wave's 16-row strip
        const unsigned short* aph = &Ahi[(wave * 16 + l16) * PITCHB];
        const unsigned short* apl = &Alo[(wave * 16 + l16) * PITCHB];
        FragBF fah, fal;
        fah.q[0] = *(const uint4*)(aph + koff0);
        fah.q[1] = *(const uint4*)(aph + koff0 + 16);
        fal.q[0] = *(const uint4*)(apl + koff0);
        fal.q[1] = *(const uint4*)(apl + koff0 + 16);
#pragma unroll
        for (int ct = 0; ct < 8; ++ct) {
            const unsigned short* bph = &Bhi[(ct * 16 + l16) * PITCHB];
            const unsigned short* bpl = &Blo[(ct * 16 + l16) * PITCHB];
            FragBF fbh, fbl;
            fbh.q[0] = *(const uint4*)(bph + koff0);
            fbh.q[1] = *(const uint4*)(bph + koff0 + 16);
            fbl.q[0] = *(const uint4*)(bpl + koff0);
            fbl.q[1] = *(const uint4*)(bpl + koff0 + 16);
            acc[ct] = __builtin_amdgcn_wmma_f32_16x16x32_bf16(
                false, fah.v, false, fbh.v, (short)0, acc[ct], false, false);
            acc[ct] = __builtin_amdgcn_wmma_f32_16x16x32_bf16(
                false, fah.v, false, fbl.v, (short)0, acc[ct], false, false);
            acc[ct] = __builtin_amdgcn_wmma_f32_16x16x32_bf16(
                false, fal.v, false, fbh.v, (short)0, acc[ct], false, false);
        }
    }

    // C/D layout: lanes 0-15 hold N=lane, M=v; lanes 16-31 hold N=lane-16, M=v+8
    const int gi0 = rowBase + wave * 16 + half * 8;  // row of element v is gi0+v
    int rlab[8];
#pragma unroll
    for (int v = 0; v < 8; ++v) rlab[v] = lab[gi0 + v];

    if (PASS == 1) {
        float rmax[8];
#pragma unroll
        for (int v = 0; v < 8; ++v) rmax[v] = -3.4e38f;
#pragma unroll
        for (int ct = 0; ct < 8; ++ct) {
            const int gj = colBase + ct * 16 + l16;
            const int clab = lab[gj];
            float cmax = -3.4e38f;
#pragma unroll
            for (int v = 0; v < 8; ++v) {
                const int gi = gi0 + v;
                float x = acc[ct][v] * 2.0f;               // /T with T=0.5
                if (rlab[v] == clab && gi != gj) x = 0.0f; // mask same-label off-diag
                rmax[v] = fmaxf(rmax[v], x);
                cmax = fmaxf(cmax, x);
            }
            cmax = fmaxf(cmax, __shfl_xor(cmax, 16, 32));  // combine both M-halves
            if (lane < 16) atomicMax(&cmaxo[gj], f2ord(cmax));
        }
#pragma unroll
        for (int v = 0; v < 8; ++v) {
            float m = rmax[v];
            m = fmaxf(m, __shfl_xor(m, 1, 32));
            m = fmaxf(m, __shfl_xor(m, 2, 32));
            m = fmaxf(m, __shfl_xor(m, 4, 32));
            m = fmaxf(m, __shfl_xor(m, 8, 32));
            if (l16 == 0) atomicMax(&rmaxo[gi0 + v], f2ord(m));
        }
    } else {
        float rmx[8], rs[8];
#pragma unroll
        for (int v = 0; v < 8; ++v) { rmx[v] = ord2f(rmaxo[gi0 + v]); rs[v] = 0.f; }
#pragma unroll
        for (int ct = 0; ct < 8; ++ct) {
            const int gj = colBase + ct * 16 + l16;
            const int clab = lab[gj];
            const float cmx = ord2f(cmaxo[gj]);
            float cs = 0.f;
#pragma unroll
            for (int v = 0; v < 8; ++v) {
                const int gi = gi0 + v;
                float x = acc[ct][v] * 2.0f;
                if (rlab[v] == clab && gi != gj) x = 0.0f;
                rs[v] += expf(x - rmx[v]);
                cs    += expf(x - cmx);
                if (gi == gj) diag[gi] = x;                // diagonal is never masked
            }
            cs += __shfl_xor(cs, 16, 32);
            if (lane < 16) atomicAdd(&csum[gj], cs);
        }
#pragma unroll
        for (int v = 0; v < 8; ++v) {
            float s = rs[v];
            s += __shfl_xor(s, 1, 32);
            s += __shfl_xor(s, 2, 32);
            s += __shfl_xor(s, 4, 32);
            s += __shfl_xor(s, 8, 32);
            if (l16 == 0) atomicAdd(&rsum[gi0 + v], s);
        }
    }
}

__global__ __launch_bounds__(256) void radsel_finalize(
    const unsigned* __restrict__ rmaxo, const unsigned* __restrict__ cmaxo,
    const float* __restrict__ rsum, const float* __restrict__ csum,
    const float* __restrict__ diag, float* __restrict__ out, int n)
{
    __shared__ float red[256];
    float a = 0.f;
    for (int i = threadIdx.x; i < n; i += 256) {
        float lser = ord2f(rmaxo[i]) + logf(rsum[i]);
        float lsec = ord2f(cmaxo[i]) + logf(csum[i]);
        a += 2.0f * diag[i] - lser - lsec;
    }
    red[threadIdx.x] = a;
    __syncthreads();
    for (int s = 128; s > 0; s >>= 1) {
        if (threadIdx.x < s) red[threadIdx.x] += red[threadIdx.x + s];
        __syncthreads();
    }
    if (threadIdx.x == 0) out[0] = -red[0] / (float)n;
}

extern "C" void kernel_launch(void* const* d_in, const int* in_sizes, int n_in,
                              void* d_out, int out_size, void* d_ws, size_t ws_size,
                              hipStream_t stream) {
    const float* F  = (const float*)d_in[0];
    const float* I  = (const float*)d_in[1];
    const int* lab  = (const int*)d_in[2];
    const int n = in_sizes[2];            // 8192
    const int d = in_sizes[0] / n;        // 256

    unsigned* rmaxo = (unsigned*)d_ws;    // n ordered-float maxes (rows)
    unsigned* cmaxo = rmaxo + n;          // n ordered-float maxes (cols)
    float* rsum = (float*)(cmaxo + n);    // n row exp-sums
    float* csum = rsum + n;               // n col exp-sums
    float* diag = csum + n;               // n diagonal values

    const int n4 = 4 * n;                 // zero rmaxo,cmaxo,rsum,csum
    radsel_init_ws<<<(n4 + 255) / 256, 256, 0, stream>>>(rmaxo, n4);

    dim3 grid(n / TILE, n / TILE);
    radsel_tile_kernel<1><<<grid, 256, 0, stream>>>(F, I, lab, rmaxo, cmaxo,
                                                    rsum, csum, diag, n, d);
    radsel_tile_kernel<2><<<grid, 256, 0, stream>>>(F, I, lab, rmaxo, cmaxo,
                                                    rsum, csum, diag, n, d);
    radsel_finalize<<<1, 256, 0, stream>>>(rmaxo, cmaxo, rsum, csum, diag,
                                           (float*)d_out, n);
}
